// Resonator_60206851555889
// MI455X (gfx1250) — matrix operand order repaired and checked by reading, not compiled, not run on hardware
//
#include <hip/hip_runtime.h>

// ---------------------------------------------------------------------------
// Resonator network on MI455X (gfx1250, wave32, WMMA + async TDM).
// All values are exactly +/-1 -> f16 WMMA (16x16x32, f32 accum) is bit-exact:
//   sim in [-2048,2048] (exact in f16), second GEMM sums < 2^24 (exact in f32).
// Codebook (f16) + transposed copy = 64MB, resident in the 192MB L2; each
// iteration streams B-matrices from L2 while the 32xKTILE A-panel is DMA'd
// into a double-buffered LDS panel by the Tensor Data Mover, overlapped with
// the WMMA compute on the other panel (TENSORcnt + workgroup barrier).
// ---------------------------------------------------------------------------

typedef __attribute__((ext_vector_type(16))) _Float16 v16h;
typedef __attribute__((ext_vector_type(8)))  _Float16 v8h;
typedef __attribute__((ext_vector_type(8)))  float    v8f;
typedef __attribute__((ext_vector_type(4)))  unsigned int v4u;
typedef __attribute__((ext_vector_type(8)))  int      v8i32;
typedef __attribute__((ext_vector_type(4)))  int      v4i32;

#define Bb 32
#define Ff 4
#define Vv 2048
#define Dd 2048
#define KTILE 512
#define NITER 50
#define PANEL_BYTES (Bb * KTILE * 2)

__device__ __forceinline__ v16h cat16(v8h lo, v8h hi) {
  return __builtin_shufflevector(lo, hi, 0,1,2,3,4,5,6,7,8,9,10,11,12,13,14,15);
}

#if __has_builtin(__builtin_amdgcn_tensor_load_to_lds)
#define HAVE_TDM 1
#else
#define HAVE_TDM 0
#endif

#if HAVE_TDM
// --- TDM: issue DMA of a 32-row x KTILE-col f16 tile (row stride F*2048
// elems) into LDS.  D# per CDNA5 ISA ch.8: group0 = {count/flags, lds_addr,
// global_addr, type=2}, group1 = {mask/data_size, tensor dims, tile dims,
// strides}; 2D tensor -> groups 2/3 zero.  Completion tracked by TENSORcnt.
__device__ __forceinline__ void tdm_issue_tile(const _Float16* gtile,
                                               unsigned ldsAddr) {
  unsigned long long ga = (unsigned long long)(uintptr_t)gtile;
  v4u g0 = { 1u,                                        // count=1, user mode
             ldsAddr,                                   // lds_addr [63:32]
             (unsigned)(ga & 0xFFFFFFFFull),            // global_addr lo
             (unsigned)((ga >> 32) & 0x01FFFFFFull)     // global_addr hi
               | 0x80000000u };                         // type=2 ("image")
  v8i32 g1 = { 0x00010000,                              // data_size=2B, mask=0
               (int)(2048u << 16),                      // tensor_dim0 = 2048
               (int)(32u   << 16),                      // tensor_dim1 = 32
               (int)((unsigned)KTILE << 16),            // tile_dim0 = KTILE
               32,                                      // tile_dim1 = 32
               Ff * 2048,                               // tensor_dim0_stride
               0, 0 };                                  // dim1_stride unused
  v4i32 z4 = { 0, 0, 0, 0 };
#if defined(__clang_major__) && (__clang_major__ >= 23)
  v8i32 z8 = { 0, 0, 0, 0, 0, 0, 0, 0 };
  __builtin_amdgcn_tensor_load_to_lds(g0, g1, z4, z4, z8, 0);
#else
  __builtin_amdgcn_tensor_load_to_lds(g0, g1, z4, z4, 0);
#endif
}

__device__ __forceinline__ void tdm_wait_all() {
#if __has_builtin(__builtin_amdgcn_s_wait_tensorcnt)
  __builtin_amdgcn_s_wait_tensorcnt(0);
#endif
}
#endif  // HAVE_TDM

// --- one-time prep ---------------------------------------------------------

__global__ __launch_bounds__(256) void res_cvt_cb(const float* __restrict__ src,
                                                  _Float16* __restrict__ dst) {
  int t = blockIdx.x * 256 + threadIdx.x;              // over F*V*D
  dst[t] = (_Float16)src[t];
}

__global__ __launch_bounds__(256) void res_cvt_cbT(const float* __restrict__ src,
                                                   _Float16* __restrict__ dst) {
  int t = blockIdx.x * 256 + threadIdx.x;              // over F*D*V
  int f = t >> 22;                                     // D*V = 2^22
  int r = t & ((1 << 22) - 1);
  int d = r >> 11;
  int v = r & (Vv - 1);
  dst[t] = (_Float16)src[((size_t)f * Vv + v) * Dd + d];
}

__global__ __launch_bounds__(256) void res_init(const float* __restrict__ src,
                                                _Float16* __restrict__ est,
                                                int* __restrict__ flags) {
  int t = blockIdx.x * 256 + threadIdx.x;              // over B*F*D
  est[t] = (_Float16)src[t];
  if (t == 0) { flags[0] = 0; flags[1] = -1; }         // done=0, k=-1
}

// --- per-iteration elementwise: new_est[b,f,d] = input*P*est (1/x==x for +-1)

__global__ __launch_bounds__(256) void res_newest(const float* __restrict__ input,
                                                  const _Float16* __restrict__ est,
                                                  _Float16* __restrict__ newEst,
                                                  int* __restrict__ flags) {
  int t = blockIdx.x * 256 + threadIdx.x;              // over B*D
  if (t == 0) flags[2] = 0;                            // changed = 0
  if (t >= Bb * Dd) return;
  int b = t >> 11, d = t & (Dd - 1);
  float e0 = (float)est[(b * Ff + 0) * Dd + d];
  float e1 = (float)est[(b * Ff + 1) * Dd + d];
  float e2 = (float)est[(b * Ff + 2) * Dd + d];
  float e3 = (float)est[(b * Ff + 3) * Dd + d];
  float P  = e0 * e1 * e2 * e3;
  float in = input[b * Dd + d] * P;
  newEst[(b * Ff + 0) * Dd + d] = (_Float16)(in * e0);
  newEst[(b * Ff + 1) * Dd + d] = (_Float16)(in * e1);
  newEst[(b * Ff + 2) * Dd + d] = (_Float16)(in * e2);
  newEst[(b * Ff + 3) * Dd + d] = (_Float16)(in * e3);
}

// --- fused WMMA GEMM: D(32x2048) = A(32x2048) x B(2048x2048)^lane-layout ---
// MODE 0: write f16 result (sim).  MODE 1: sign -> next estimate + changed
// flag, with done-freeze copy path.  A panels DMA'd into double-buffered LDS
// by the TDM, overlapped with compute.

template<int MODE>
__global__ __launch_bounds__(256)
void resonator_gemm(const _Float16* __restrict__ Amat,   // rows at (b*F+f)*2048
                    const _Float16* __restrict__ Bmat,   // rows at (f*2048+n)*2048
                    _Float16* __restrict__ Out,          // (b*F+f)*2048 + n
                    const _Float16* __restrict__ estCur, // MODE 1 only
                    int* __restrict__ flags) {
  const int f    = blockIdx.y;
  const int n0b  = blockIdx.x * 256;                    // 8 waves x 32 cols
  const int lane = threadIdx.x & 31;
  const int wave = threadIdx.x >> 5;

  if (MODE == 1) {
    if (flags[0]) {                                     // done: freeze estimate
      for (int j = 0; j < 32; ++j) {
        int e = (int)threadIdx.x + j * 256;             // 0..8191
        int b = e >> 8;
        int n = n0b + (e & 255);
        int idx = (b * Ff + f) * 2048 + n;
        Out[idx] = estCur[idx];
      }
      return;
    }
  }

  __shared__ _Float16 lds[2 * Bb * KTILE];              // double-buffered panel

  const int hi  = lane >> 4;                            // K-half select
  const int l15 = lane & 15;
  const int nw  = n0b + wave * 32;                      // 2 N-strips per wave

  v8f c00 = {}; v8f c01 = {}; v8f c10 = {}; v8f c11 = {};

  // B lane layout (16x16x32 f16): lane holds K chunks k0+8*hi..+7 and
  // k0+16+8*hi..+7 of col n0+(lane&15) -> two 16B chunks / lane.
  const _Float16* bbase0 = Bmat + ((size_t)(f * 2048 + nw      + l15)) * 2048 + hi * 8;
  const _Float16* bbase1 = Bmat + ((size_t)(f * 2048 + nw + 16 + l15)) * 2048 + hi * 8;
  const _Float16* Atile  = Amat + (size_t)f * 2048;

#if HAVE_TDM
  const unsigned ldsBase = (unsigned)(uintptr_t)(&lds[0]);
  // Prologue: DMA panel 0, wait, publish to all waves.
  if (wave == 0) {
    tdm_issue_tile(Atile, ldsBase);
    tdm_wait_all();
  }
  __syncthreads();
#endif

  int p = 0;
  for (int kc = 0; kc < 2048; kc += KTILE) {
#if HAVE_TDM
    // Kick off the next panel's DMA into the other buffer; it completes
    // behind this chunk's 64 WMMAs + B-stream (waited at loop bottom).
    if (wave == 0 && kc + KTILE < 2048) {
      tdm_issue_tile(Atile + kc + KTILE,
                     ldsBase + (unsigned)((p ^ 1) * PANEL_BYTES));
    }
#else
    // Fallback: cooperative VGPR copy of the A panel (single-buffered).
    for (int j = 0; j < 8; ++j) {
      int t  = (int)threadIdx.x + j * 256;              // 0..2047
      int m  = t >> 6;                                  // 64 v8h per row
      int c8 = (t & 63) << 3;
      lds[p * Bb * KTILE + m * KTILE + c8 + 0] = Atile[((size_t)m * Ff) * 2048 + kc + c8 + 0];
      *(v8h*)&lds[p * Bb * KTILE + m * KTILE + c8] =
          *(const v8h*)&Atile[((size_t)m * Ff) * 2048 + kc + c8];
    }
    __syncthreads();
#endif

    if (kc + KTILE < 2048) {                            // L2 stream-ahead
      __builtin_prefetch(bbase0 + kc + KTILE, 0, 1);
      __builtin_prefetch(bbase1 + kc + KTILE, 0, 1);
    }

    const _Float16* panel = &lds[p * Bb * KTILE];
    #pragma unroll 8
    for (int k = 0; k < KTILE; k += 32) {
      // A lane layout: chunks at k + 8*hi and k + 16 + 8*hi (16B each)
      const _Float16* ar0 = &panel[(l15     ) * KTILE + k + hi * 8];
      const _Float16* ar1 = &panel[(l15 + 16) * KTILE + k + hi * 8];
      v16h a0 = cat16(*(const v8h*)(ar0), *(const v8h*)(ar0 + 16));
      v16h a1 = cat16(*(const v8h*)(ar1), *(const v8h*)(ar1 + 16));
      const _Float16* bp0 = bbase0 + kc + k;
      const _Float16* bp1 = bbase1 + kc + k;
      v16h b0 = cat16(*(const v8h*)(bp0), *(const v8h*)(bp0 + 16));
      v16h b1 = cat16(*(const v8h*)(bp1), *(const v8h*)(bp1 + 16));
      c00 = __builtin_amdgcn_wmma_f32_16x16x32_f16(false, a0, false, b0, (short)0, c00, false, false);
      c01 = __builtin_amdgcn_wmma_f32_16x16x32_f16(false, a0, false, b1, (short)0, c01, false, false);
      c10 = __builtin_amdgcn_wmma_f32_16x16x32_f16(false, a1, false, b0, (short)0, c10, false, false);
      c11 = __builtin_amdgcn_wmma_f32_16x16x32_f16(false, a1, false, b1, (short)0, c11, false, false);
    }

#if HAVE_TDM
    if (wave == 0 && kc + KTILE < 2048) tdm_wait_all(); // next panel landed
#endif
    __syncthreads();
    p ^= 1;
  }

  // C/D layout: VGPR r -> (M = r + 8*hi, N = lane&15)
  bool anyDiff = false;
  #pragma unroll
  for (int r = 0; r < 8; ++r) {
    int m  = r + hi * 8;
    float vals[4] = { c00[r], c01[r], c10[r], c11[r] };
    int   bs[4]   = { m, m, 16 + m, 16 + m };
    int   ns[4]   = { nw + l15, nw + 16 + l15, nw + l15, nw + 16 + l15 };
    #pragma unroll
    for (int q = 0; q < 4; ++q) {
      size_t idx = ((size_t)(bs[q] * Ff + f)) * 2048 + ns[q];
      if (MODE == 0) {
        Out[idx] = (_Float16)vals[q];                   // exact integer <=2048
      } else {
        _Float16 s = (vals[q] >= 0.0f) ? (_Float16)1.0f : (_Float16)-1.0f;
        if ((float)estCur[idx] != (float)s) anyDiff = true;
        Out[idx] = s;
      }
    }
  }
  if (MODE == 1 && anyDiff) atomicOr(&flags[2], 1);
}

// --- scalar control: k/done per reference scan semantics -------------------

__global__ void res_commit(int* __restrict__ flags) {
  if (!flags[0]) {
    flags[1] += 1;                                      // k++
    if (flags[2] == 0) flags[0] = 1;                    // converged -> done
  }
}

// --- finalize: argmax|sim| and output writeback ----------------------------

__global__ __launch_bounds__(128) void res_argmax(const _Float16* __restrict__ sim,
                                                  float* __restrict__ outcome) {
  int t = threadIdx.x;                                  // 0..127 == b*F+f
  const _Float16* s = sim + (size_t)t * Vv;
  float best = -1.0f; int bi = 0;
  for (int v = 0; v < Vv; ++v) {
    float a = fabsf((float)s[v]);
    if (a > best) { best = a; bi = v; }                 // first-max like jnp
  }
  outcome[t] = (float)bi;
}

__global__ __launch_bounds__(256) void res_writeout(const _Float16* __restrict__ est,
                                                    float* __restrict__ out,
                                                    const int* __restrict__ flags) {
  int t = blockIdx.x * 256 + threadIdx.x;               // over B*F*D
  out[t] = (float)est[t];
  if (t == 0) out[Bb * Ff * Dd + Bb * Ff] = (float)flags[1];  // k
}

// ---------------------------------------------------------------------------

extern "C" void kernel_launch(void* const* d_in, const int* in_sizes, int n_in,
                              void* d_out, int out_size, void* d_ws, size_t ws_size,
                              hipStream_t stream) {
  const float* input   = (const float*)d_in[0];   // (B,D)
  const float* initEst = (const float*)d_in[1];   // (B,F,D)
  const float* cb      = (const float*)d_in[2];   // (F,V,D)
  float* out = (float*)d_out;
  char*  ws  = (char*)d_ws;

  _Float16* cb16   = (_Float16*)(ws);                          // 32 MB
  _Float16* cbT16  = (_Float16*)(ws + (size_t)33554432);       // 32 MB
  _Float16* estA   = (_Float16*)(ws + (size_t)67108864);       // 512 KB
  _Float16* estB   = (_Float16*)(ws + (size_t)67633152);       // 512 KB
  _Float16* newEst = (_Float16*)(ws + (size_t)68157440);       // 512 KB
  _Float16* sim16  = (_Float16*)(ws + (size_t)68681728);       // 512 KB
  int*      flags  = (int*)     (ws + (size_t)69206016);       // done,k,changed
  _Float16* est[2] = { estA, estB };

  dim3 blk(256);
  res_cvt_cb <<<(Ff * Vv * Dd) / 256, blk, 0, stream>>>(cb, cb16);
  res_cvt_cbT<<<(Ff * Dd * Vv) / 256, blk, 0, stream>>>(cb, cbT16);
  res_init   <<<(Bb * Ff * Dd) / 256, blk, 0, stream>>>(initEst, est[0], flags);

  dim3 gg(2048 / 256, Ff);
  for (int i = 0; i < NITER; ++i) {
    _Float16* cur = est[i & 1];
    _Float16* nxt = est[(i + 1) & 1];
    res_newest<<<(Bb * Dd) / 256, blk, 0, stream>>>(input, cur, newEst, flags);
    resonator_gemm<0><<<gg, blk, 0, stream>>>(newEst, cb16,  sim16, nullptr, flags);
    resonator_gemm<1><<<gg, blk, 0, stream>>>(sim16,  cbT16, nxt,   cur,     flags);
    res_commit<<<1, 1, 0, stream>>>(flags);
  }

  _Float16* fin = est[NITER & 1];
  resonator_gemm<0><<<gg, blk, 0, stream>>>(fin, cb16, sim16, nullptr, flags);
  res_argmax  <<<1, 128, 0, stream>>>(sim16, out + (size_t)Bb * Ff * Dd);
  res_writeout<<<(Bb * Ff * Dd) / 256, blk, 0, stream>>>(fin, out, flags);
}
